// Net_24395414241708
// MI455X (gfx1250) — matrix operand order repaired and checked by reading, not compiled, not run on hardware
//
#include <hip/hip_runtime.h>
#include <hip/hip_bf16.h>

#define NN 10000      // nodes
#define NE 160000     // edges
#define F  128        // feature width (F0 == H == 128)
#define ED 16         // edge_attr dim

typedef __attribute__((ext_vector_type(16))) __bf16 v16bf;
typedef __attribute__((ext_vector_type(8)))  float  v8f;

union Frag { uint4 q[2]; v16bf v; };

__device__ __forceinline__ v8f wmma_bf16(const Frag& a, const Frag& b, v8f c) {
  return __builtin_amdgcn_wmma_f32_16x16x32_bf16(false, a.v, false, b.v, (short)0, c, false, false);
}

__device__ __forceinline__ void atomicMaxF(float* addr, float val) {
  unsigned int* ua = (unsigned int*)addr;
  unsigned int cur = *ua;
  while (__uint_as_float(cur) < val) {
    unsigned int prev = atomicCAS(ua, cur, __float_as_uint(val));
    if (prev == cur) break;
    cur = prev;
  }
}
__device__ __forceinline__ void atomicMinF(float* addr, float val) {
  unsigned int* ua = (unsigned int*)addr;
  unsigned int cur = *ua;
  while (__uint_as_float(cur) > val) {
    unsigned int prev = atomicCAS(ua, cur, __float_as_uint(val));
    if (prev == cur) break;
    cur = prev;
  }
}

// ---------------------------------------------------------------- utilities
__global__ void k_fill(float* p, long n, float v) {
  for (long i = blockIdx.x * (long)blockDim.x + threadIdx.x; i < n;
       i += (long)gridDim.x * blockDim.x) p[i] = v;
}

// convert fp32 [batch][K][Nc] -> bf16 transposed [batch][Nc][K]
__global__ void k_cvtT(const float* __restrict__ src, __bf16* __restrict__ dst,
                       int K, int Nc, int batch) {
  long kn = (long)K * Nc, total = kn * batch;
  for (long i = blockIdx.x * (long)blockDim.x + threadIdx.x; i < total;
       i += (long)gridDim.x * blockDim.x) {
    int b = (int)(i / kn);
    long r = i - (long)b * kn;
    int k = (int)(r / Nc);
    int n = (int)(r - (long)k * Nc);
    dst[((size_t)b * Nc + n) * K + k] = (__bf16)src[i];
  }
}

__global__ void k_deg(const int* __restrict__ ei, float* __restrict__ deg) {
  int e = blockIdx.x * blockDim.x + threadIdx.x;
  if (e < NE) atomicAdd(&deg[ei[NE + e]], 1.0f);
}

__global__ void k_avglog(const float* __restrict__ deg, float* __restrict__ out) {
  __shared__ float red[256];
  float s = 0.f;
  for (int i = threadIdx.x; i < NN; i += 256) s += logf(deg[i] + 1.0f);
  red[threadIdx.x] = s;
  __syncthreads();
  for (int o = 128; o > 0; o >>= 1) {
    if (threadIdx.x < o) red[threadIdx.x] += red[threadIdx.x + o];
    __syncthreads();
  }
  if (threadIdx.x == 0) out[0] = red[0] / (float)NN;
}

// --------------------------------------------- fused edge encoder + pre-NN
// m[e, t*F+f] = bpre + [x[dst], x[src], edge_attr@We + be] @ Wpre[t]
__global__ void __launch_bounds__(128)
k_edge_pre(const float* __restrict__ x, const int* __restrict__ ei,
           const float* __restrict__ ea, const float* __restrict__ We,
           const float* __restrict__ be, const __bf16* __restrict__ BpreT,
           const float* __restrict__ bpre, float* __restrict__ m, int T) {
  __shared__ __align__(16) __bf16 A[16][392];   // 16 edges x 384 K (bf16)
  __shared__ int sdst[16], ssrc[16];
  const int tid = threadIdx.x;
  const int etile = blockIdx.x * 16;
  if (tid < 16) {
    ssrc[tid] = ei[etile + tid];
    sdst[tid] = ei[NE + etile + tid];
  }
  __syncthreads();
  // Build A tile: cols 0..127 = x[dst], 128..255 = x[src], 256..383 = encoder
  for (int idx = tid; idx < 16 * 384; idx += 128) {
    int r = idx / 384;
    int c = idx - r * 384;
    float v;
    if (c < 128) {
      v = x[(size_t)sdst[r] * F + c];
    } else if (c < 256) {
      v = x[(size_t)ssrc[r] * F + (c - 128)];
    } else {
      int c2 = c - 256;
      const float* row = ea + (size_t)(etile + r) * ED;
      float s = be[c2];
#pragma unroll
      for (int j = 0; j < ED; ++j) s += row[j] * We[j * F + c2];
      v = s;
    }
    A[r][c] = (__bf16)v;
  }
  __syncthreads();
  const int wave = tid >> 5, lane = tid & 31;
  const int ln = lane & 15, hi = lane >> 4;
  const int TF = T * F;
  const int nTiles = TF >> 4;
  for (int tile = wave; tile < nTiles; tile += 4) {
    const int nb = tile * 16;
    v8f acc = {};
    const __bf16* brow = BpreT + (size_t)(nb + ln) * 384 + 8 * hi;
#pragma unroll 4
    for (int k0 = 0; k0 < 384; k0 += 32) {
      Frag a, b;
      a.q[0] = *(const uint4*)&A[ln][k0 + 8 * hi];
      a.q[1] = *(const uint4*)&A[ln][k0 + 16 + 8 * hi];
      b.q[0] = ((const uint4*)(brow + k0))[0];
      b.q[1] = ((const uint4*)(brow + k0 + 16))[0];
      acc = wmma_bf16(a, b, acc);
    }
    const int col = nb + ln;
    const float bias = bpre[col];
#pragma unroll
    for (int r = 0; r < 8; ++r) {
      int row = r + 8 * hi;
      m[(size_t)(etile + row) * TF + col] = acc[r] + bias;
    }
  }
}

// ----------------------------------------------------- segment aggregation
__global__ void k_agg(const float* __restrict__ m, const int* __restrict__ ei,
                      float* s1, float* s2, float* mn, float* mx, int lgTF) {
  const long total = (long)NE << lgTF;
  const int TFm = (1 << lgTF) - 1;
  for (long i = blockIdx.x * (long)blockDim.x + threadIdx.x; i < total;
       i += (long)gridDim.x * blockDim.x) {
    int e = (int)(i >> lgTF);
    int c = (int)i & TFm;
    int d = ei[NE + e];
    float v = m[i];
    size_t o = ((size_t)d << lgTF) + c;
    atomicAdd(&s1[o], v);
    atomicAdd(&s2[o], v * v);
    atomicMaxF(&mx[o], v);
    atomicMinF(&mn[o], v);
  }
}

// ------------------------------- scalers + post-NN (13F = 1664 wide, WMMA)
__global__ void __launch_bounds__(128)
k_post(const float* __restrict__ x, const float* __restrict__ s1,
       const float* __restrict__ s2, const float* __restrict__ mn,
       const float* __restrict__ mx, const float* __restrict__ deg,
       const float* __restrict__ avg_log, const __bf16* __restrict__ BpostT,
       const float* __restrict__ bpost, float* __restrict__ mpost,
       int T, int FOUT) {
  const int KT = 13 * F;  // 1664
  __shared__ __align__(16) __bf16 A[16][1672];
  __shared__ float sdegc[16], samp[16], satt[16], shas[16];
  const int tid = threadIdx.x;
  const int t = blockIdx.y;
  const int ntile = blockIdx.x * 16;
  const float al = avg_log[0];
  if (tid < 16) {
    float d = deg[ntile + tid];
    float dc = fmaxf(d, 1.0f);
    float logd = logf(dc + 1.0f);
    sdegc[tid] = dc;
    samp[tid] = logd / al;
    satt[tid] = al / logd;
    shas[tid] = (d > 0.f) ? 1.f : 0.f;
  }
  __syncthreads();
  const int TF = T * F;
  for (int idx = tid; idx < 16 * KT; idx += 128) {
    int r = idx / KT;
    int c = idx - r * KT;
    int node = ntile + r;
    float v;
    if (c < 128) {
      v = x[(size_t)node * F + c];
    } else {
      int cc = c - 128;
      int group = cc / 512;            // 0:identity 1:amplify 2:attenuate
      int a = cc - group * 512;        // within [mean|min|max|std] x 128
      int which = a >> 7;
      int f = a & 127;
      size_t o = (size_t)node * TF + t * F + f;
      float dc = sdegc[r];
      float mean = s1[o] / dc;
      float val;
      if (which == 0)      val = mean;
      else if (which == 1) val = (shas[r] != 0.f) ? mn[o] : 0.f;
      else if (which == 2) val = (shas[r] != 0.f) ? mx[o] : 0.f;
      else {
        float var = s2[o] / dc - mean * mean;
        val = sqrtf(fmaxf(var, 0.f) + 1e-5f);
      }
      float scale = (group == 0) ? 1.f : ((group == 1) ? samp[r] : satt[r]);
      v = val * scale;
    }
    A[r][c] = (__bf16)v;
  }
  __syncthreads();
  const int wave = tid >> 5, lane = tid & 31;
  const int ln = lane & 15, hi = lane >> 4;
  const int nTiles = FOUT >> 4;
  for (int tile = wave; tile < nTiles; tile += 4) {
    const int nb = tile * 16;
    v8f acc = {};
    const __bf16* brow = BpostT + ((size_t)t * FOUT + nb + ln) * KT + 8 * hi;
#pragma unroll 4
    for (int k0 = 0; k0 < KT; k0 += 32) {
      Frag a, b;
      a.q[0] = *(const uint4*)&A[ln][k0 + 8 * hi];
      a.q[1] = *(const uint4*)&A[ln][k0 + 16 + 8 * hi];
      b.q[0] = ((const uint4*)(brow + k0))[0];
      b.q[1] = ((const uint4*)(brow + k0 + 16))[0];
      acc = wmma_bf16(a, b, acc);
    }
    const int col = nb + ln;
    const float bias = bpost[t * FOUT + col];
#pragma unroll
    for (int r = 0; r < 8; ++r) {
      int row = r + 8 * hi;
      mpost[(size_t)(ntile + row) * (T * FOUT) + t * FOUT + col] = acc[r] + bias;
    }
  }
}

// --------------------------------------- final per-layer linear (+ ReLU)
__global__ void __launch_bounds__(128)
k_lin(const float* __restrict__ in, const __bf16* __restrict__ WT,
      const float* __restrict__ bias, float* __restrict__ out) {
  __shared__ __align__(16) __bf16 A[16][136];
  const int tid = threadIdx.x;
  const int ntile = blockIdx.x * 16;
  for (int idx = tid; idx < 16 * 128; idx += 128) {
    int r = idx >> 7, c = idx & 127;
    A[r][c] = (__bf16)in[(size_t)(ntile + r) * F + c];
  }
  __syncthreads();
  const int wave = tid >> 5, lane = tid & 31;
  const int ln = lane & 15, hi = lane >> 4;
  for (int tile = wave; tile < 8; tile += 4) {
    const int nb = tile * 16;
    v8f acc = {};
    const __bf16* brow = WT + (size_t)(nb + ln) * 128 + 8 * hi;
#pragma unroll
    for (int k0 = 0; k0 < 128; k0 += 32) {
      Frag a, b;
      a.q[0] = *(const uint4*)&A[ln][k0 + 8 * hi];
      a.q[1] = *(const uint4*)&A[ln][k0 + 16 + 8 * hi];
      b.q[0] = ((const uint4*)(brow + k0))[0];
      b.q[1] = ((const uint4*)(brow + k0 + 16))[0];
      acc = wmma_bf16(a, b, acc);
    }
    const int col = nb + ln;
    const float bs = bias[col];
#pragma unroll
    for (int r = 0; r < 8; ++r) {
      float v = fmaxf(acc[r] + bs, 0.f);
      out[(size_t)(ntile + r + 8 * hi) * F + col] = v;
    }
  }
}

// ------------------------------------------- tail MLP 128->46->16 + logsmx
__global__ void k_mlp(const float* __restrict__ x2, const float* __restrict__ W0,
                      const float* __restrict__ b0, const float* __restrict__ W1,
                      const float* __restrict__ b1, float* __restrict__ out) {
  int n = blockIdx.x * blockDim.x + threadIdx.x;
  if (n >= NN) return;
  const float* xr = x2 + (size_t)n * F;
  float h[46];
  for (int j = 0; j < 46; ++j) h[j] = b0[j];
  for (int k = 0; k < F; ++k) {
    float xv = xr[k];
    const float* wr = W0 + (size_t)k * 46;
    for (int j = 0; j < 46; ++j) h[j] += xv * wr[j];
  }
  for (int j = 0; j < 46; ++j) h[j] = fmaxf(h[j], 0.f);
  float lg[16];
  for (int c = 0; c < 16; ++c) lg[c] = b1[c];
  for (int j = 0; j < 46; ++j) {
    float hv = h[j];
    const float* wr = W1 + (size_t)j * 16;
    for (int c = 0; c < 16; ++c) lg[c] += hv * wr[c];
  }
  float mxv = -3.0e38f;
  for (int c = 0; c < 16; ++c) mxv = fmaxf(mxv, lg[c]);
  float lse = 0.f;
  for (int c = 0; c < 16; ++c) lse += expf(lg[c] - mxv);
  lse = logf(lse) + mxv;
  for (int c = 0; c < 16; ++c) out[(size_t)n * 16 + c] = lg[c] - lse;
}

// ----------------------------------------------------------------- driver
extern "C" void kernel_launch(void* const* d_in, const int* in_sizes, int n_in,
                              void* d_out, int out_size, void* d_ws, size_t ws_size,
                              hipStream_t stream) {
  (void)in_sizes; (void)n_in; (void)out_size; (void)ws_size;
  const float* x0   = (const float*)d_in[0];
  const int*   ei   = (const int*)d_in[1];
  const float* ea   = (const float*)d_in[2];
  const float* We0  = (const float*)d_in[3];
  const float* be0  = (const float*)d_in[4];
  const float* Wpre0  = (const float*)d_in[5];
  const float* bpre0  = (const float*)d_in[6];
  const float* Wpost0 = (const float*)d_in[7];
  const float* bpost0 = (const float*)d_in[8];
  const float* Wlin0  = (const float*)d_in[9];
  const float* blin0  = (const float*)d_in[10];
  const float* We1  = (const float*)d_in[11];
  const float* be1  = (const float*)d_in[12];
  const float* Wpre1  = (const float*)d_in[13];
  const float* bpre1  = (const float*)d_in[14];
  const float* Wpost1 = (const float*)d_in[15];
  const float* bpost1 = (const float*)d_in[16];
  const float* Wlin1  = (const float*)d_in[17];
  const float* blin1  = (const float*)d_in[18];
  const float* Wm0 = (const float*)d_in[19];
  const float* bm0 = (const float*)d_in[20];
  const float* Wm1 = (const float*)d_in[21];
  const float* bm1 = (const float*)d_in[22];
  float* outp = (float*)d_out;

  char* w = (char*)d_ws;
  size_t off = 0;
  auto take = [&](size_t bytes) -> void* {
    void* p = w + off;
    off = (off + bytes + 255) & ~(size_t)255;
    return p;
  };
  float* deg  = (float*)take((size_t)NN * 4);
  float* avl  = (float*)take(4);
  float* s1   = (float*)take((size_t)NN * 256 * 4);
  float* s2   = (float*)take((size_t)NN * 256 * 4);
  float* mnb  = (float*)take((size_t)NN * 256 * 4);
  float* mxb  = (float*)take((size_t)NN * 256 * 4);
  float* mpost = (float*)take((size_t)NN * F * 4);
  float* x1   = (float*)take((size_t)NN * F * 4);
  float* x2   = (float*)take((size_t)NN * F * 4);
  float* mbuf = (float*)take((size_t)NE * 256 * 4);
  __bf16* wpreT  = (__bf16*)take((size_t)2 * 128 * 384 * 2);
  __bf16* wpostT = (__bf16*)take((size_t)128 * 1664 * 2);
  __bf16* wlinT  = (__bf16*)take((size_t)128 * 128 * 2);

  // degree + avg_log (shared by both layers)
  k_fill<<<256, 256, 0, stream>>>(deg, NN, 0.f);
  k_deg<<<(NE + 255) / 256, 256, 0, stream>>>(ei, deg);
  k_avglog<<<1, 256, 0, stream>>>(deg, avl);

  // ------------------------------ layer 0 (T=1, FOUT=128)
  k_cvtT<<<512, 256, 0, stream>>>(Wpre0, wpreT, 384, 128, 1);
  k_cvtT<<<1024, 256, 0, stream>>>(Wpost0, wpostT, 1664, 128, 1);
  k_cvtT<<<256, 256, 0, stream>>>(Wlin0, wlinT, 128, 128, 1);
  k_fill<<<2048, 256, 0, stream>>>(s1, (long)NN * 128, 0.f);
  k_fill<<<2048, 256, 0, stream>>>(s2, (long)NN * 128, 0.f);
  k_fill<<<2048, 256, 0, stream>>>(mnb, (long)NN * 128, 3.0e38f);
  k_fill<<<2048, 256, 0, stream>>>(mxb, (long)NN * 128, -3.0e38f);
  k_edge_pre<<<NE / 16, 128, 0, stream>>>(x0, ei, ea, We0, be0, wpreT, bpre0, mbuf, 1);
  k_agg<<<4096, 256, 0, stream>>>(mbuf, ei, s1, s2, mnb, mxb, 7);
  k_post<<<dim3(NN / 16, 1), 128, 0, stream>>>(x0, s1, s2, mnb, mxb, deg, avl,
                                               wpostT, bpost0, mpost, 1, 128);
  k_lin<<<NN / 16, 128, 0, stream>>>(mpost, wlinT, blin0, x1);

  // ------------------------------ layer 1 (T=2, FOUT=64)
  k_cvtT<<<512, 256, 0, stream>>>(Wpre1, wpreT, 384, 128, 2);
  k_cvtT<<<1024, 256, 0, stream>>>(Wpost1, wpostT, 1664, 64, 2);
  k_cvtT<<<256, 256, 0, stream>>>(Wlin1, wlinT, 128, 128, 1);
  k_fill<<<2048, 256, 0, stream>>>(s1, (long)NN * 256, 0.f);
  k_fill<<<2048, 256, 0, stream>>>(s2, (long)NN * 256, 0.f);
  k_fill<<<2048, 256, 0, stream>>>(mnb, (long)NN * 256, 3.0e38f);
  k_fill<<<2048, 256, 0, stream>>>(mxb, (long)NN * 256, -3.0e38f);
  k_edge_pre<<<NE / 16, 128, 0, stream>>>(x1, ei, ea, We1, be1, wpreT, bpre1, mbuf, 2);
  k_agg<<<4096, 256, 0, stream>>>(mbuf, ei, s1, s2, mnb, mxb, 8);
  k_post<<<dim3(NN / 16, 2), 128, 0, stream>>>(x1, s1, s2, mnb, mxb, deg, avl,
                                               wpostT, bpost1, mpost, 2, 64);
  k_lin<<<NN / 16, 128, 0, stream>>>(mpost, wlinT, blin1, x2);

  // ------------------------------ tail MLP + log-softmax
  k_mlp<<<(NN + 127) / 128, 128, 0, stream>>>(x2, Wm0, bm0, Wm1, bm1, outp);
}
